// BiCPRNN_11467562680760
// MI455X (gfx1250) — compile-verified, matrix operand order checked by reading, not run on hardware
//
#include <hip/hip_runtime.h>
#include <hip/hip_bf16.h>
#include <stddef.h>

// ---------------------------------------------------------------------------
// BiCPRNN on MI455X (gfx1250): bf16 WMMA everywhere, f32 accumulate.
//   VOCAB=10000 INPUT=256 HIDDEN=512 RANK=64 BATCH=64 SEQ=256
// ---------------------------------------------------------------------------

typedef __bf16 bf16_t;
typedef __attribute__((ext_vector_type(16))) __bf16 v16bf;
typedef __attribute__((ext_vector_type(8)))  __bf16 v8bf;
typedef __attribute__((ext_vector_type(8)))  float  v8f;

#define VOCAB  10000
#define INPUT  256
#define HIDDEN 512
#define RANK   64
#define BATCH  64
#define SEQ    256

__device__ __forceinline__ v8f wmma_bf16(v16bf a, v16bf b, v8f c) {
    return __builtin_amdgcn_wmma_f32_16x16x32_bf16(
        /*neg_a=*/false, a, /*neg_b=*/false, b,
        /*c_mod=*/(short)0, c, /*reuse_a=*/false, /*reuse_b=*/false);
}

// Fragment for 16x32 bf16 A-tile (or KxN B-tile stored [N][K]):
// lane row = lane&15 ; K-half base kb = (lane>=16)?8:0 folded into `row`
// pointer by caller. Elements [0..7] = K k+0..7, [8..15] = K k+16..23.
union U16 { v16bf v; v8bf h[2]; };

__device__ __forceinline__ v16bf load_frag(const bf16_t* row, int k) {
    U16 u;
    u.h[0] = *(const v8bf*)(row + k);
    u.h[1] = *(const v8bf*)(row + k + 16);
    return u.v;
}

__device__ __forceinline__ v16bf load_frag_f32(const float* row, int k) {
    v16bf f;
#pragma unroll
    for (int i = 0; i < 8; ++i) {
        f[i]     = (bf16_t)row[k + i];
        f[i + 8] = (bf16_t)row[k + 16 + i];
    }
    return f;
}

__device__ __forceinline__ v16bf zero_frag() {
    v16bf z = {};
    return z;
}

// ---------------------------------------------------------------------------
// Kernel 1: f32 -> bf16 elementwise (emb, W_dec)
// ---------------------------------------------------------------------------
__global__ void k_f32_to_bf16(const float* __restrict__ src,
                              bf16_t* __restrict__ dst, long n) {
    long i = (long)blockIdx.x * blockDim.x + threadIdx.x;
    long stride = (long)gridDim.x * blockDim.x;
    for (; i < n; i += stride) dst[i] = (bf16_t)src[i];
}

// Kernel 2: Bt[r][i] = B[i][r]  (B is [INPUT][RANK]) -> bf16, [RANK][INPUT]
__global__ void k_make_Bt(const float* __restrict__ B,
                          bf16_t* __restrict__ Bt) {
    int idx = blockIdx.x * blockDim.x + threadIdx.x;   // < RANK*INPUT
    if (idx < RANK * INPUT) {
        int r = idx / INPUT, i = idx % INPUT;
        Bt[idx] = (bf16_t)B[i * RANK + r];
    }
}

// ---------------------------------------------------------------------------
// Kernel 3: Bx[s*64+b][r] = emb[inp[b][s]] . B[:,r]   (gathered WMMA GEMM)
//   grid: 256 blocks (one per 64 rows), 128 threads (4 waves).
//   wave w owns rows m0..m0+15 and all 4 N-tiles of RANK=64.
// ---------------------------------------------------------------------------
__global__ void k_bx(const int* __restrict__ inp,
                     const bf16_t* __restrict__ embb,
                     const bf16_t* __restrict__ Bt,
                     float* __restrict__ Bx) {
    const int w    = threadIdx.x >> 5;
    const int lane = threadIdx.x & 31;
    const int rr   = lane & 15;
    const int kb   = ((lane >> 4) & 1) * 8;
    const int m0   = blockIdx.x * 64 + w * 16;

    const int row = m0 + rr;            // row in [S*B), ordered [S][B]
    const int s   = row >> 6;
    const int b   = row & 63;
    const int tok = inp[b * SEQ + s];
    const bf16_t* arow = embb + (size_t)tok * INPUT + kb;

    v8f acc[4] = {{}, {}, {}, {}};
#pragma unroll
    for (int kt = 0; kt < INPUT / 32; ++kt) {
        const int k0 = kt * 32;
        v16bf a = load_frag(arow, k0);
#pragma unroll
        for (int ni = 0; ni < 4; ++ni) {
            const bf16_t* brow = Bt + (size_t)(ni * 16 + rr) * INPUT + kb;
            acc[ni] = wmma_bf16(a, load_frag(brow, k0), acc[ni]);
        }
    }
    const int mbase = m0 + (((lane >> 4) & 1) ? 8 : 0);
#pragma unroll
    for (int ni = 0; ni < 4; ++ni) {
        const int ncol = ni * 16 + rr;
#pragma unroll
        for (int r = 0; r < 8; ++r)
            Bx[(size_t)(mbase + r) * RANK + ncol] = acc[ni][r];
    }
}

// ---------------------------------------------------------------------------
// Kernel 4: the serial recurrence, one WGP, whole state in 266 KB of LDS.
//   h_new = tanh(((h@A) * Bx[s]) @ C^T + d)
//   1024 threads = 32 waves. Phase1: waves 0..15 each do one 16x16 tile of
//   h@A (K=512, 16 WMMAs) then multiply by Bx and stage bf16 tile T.
//   Phase2: wave w does column-block n0=w*16 of T@C^T (M=64,K=64), tanh+d,
//   writes h (f32, LDS) and hidden_seq (bf16, global).
// ---------------------------------------------------------------------------
__global__ __launch_bounds__(1024)
void k_recur(const float* __restrict__ Ag, const float* __restrict__ Cg,
             const float* __restrict__ dg, const float* __restrict__ Bx,
             bf16_t* __restrict__ Hb, float* __restrict__ hout) {
    extern __shared__ char smem[];
    float*  h  = (float*)smem;                               // [64][512] 128KB
    bf16_t* At = (bf16_t*)(smem + 131072);                   // [64][512] At[r][k]=A[k][r]
    bf16_t* Cb = (bf16_t*)(smem + 131072 + 65536);           // [512][64] = C rows
    bf16_t* T  = (bf16_t*)(smem + 131072 + 2 * 65536);       // [64][64]
    float*  dl = (float*)(smem + 131072 + 2 * 65536 + 8192); // [512]

    const int tid  = threadIdx.x;
    const int w    = tid >> 5;
    const int lane = tid & 31;
    const int rr   = lane & 15;
    const int kb   = ((lane >> 4) & 1) * 8;
    const int mh   = ((lane >> 4) & 1) ? 8 : 0;

    for (int i = tid; i < RANK * HIDDEN; i += 1024) {
        int r = i >> 9, k = i & 511;
        At[i] = (bf16_t)Ag[k * RANK + r];           // A is [HIDDEN][RANK]
        h[i]  = 0.0f;                               // h0 = zeros (same extent)
    }
    for (int i = tid; i < HIDDEN * RANK; i += 1024)
        Cb[i] = (bf16_t)Cg[i];                      // C rows are already [n][k]
    if (tid < HIDDEN) dl[tid] = dg[tid];
    __syncthreads();

    for (int s = 0; s < SEQ; ++s) {
        // ---- phase 1: hA = h @ A, times bx, into T (bf16) ----
        if (w < 16) {
            const int m0 = (w >> 2) * 16, n0 = (w & 3) * 16;
            const float*  arow = h  + (m0 + rr) * HIDDEN + kb;
            const bf16_t* brow = At + (n0 + rr) * HIDDEN + kb;
            v8f acc = {};
#pragma unroll
            for (int kt = 0; kt < HIDDEN / 32; ++kt) {
                acc = wmma_bf16(load_frag_f32(arow, kt * 32),
                                load_frag(brow, kt * 32), acc);
            }
            const int ncol = n0 + rr, mbase = m0 + mh;
#pragma unroll
            for (int r = 0; r < 8; ++r) {
                int m = mbase + r;
                float bx = Bx[(size_t)(s * BATCH + m) * RANK + ncol];
                T[m * RANK + ncol] = (bf16_t)(acc[r] * bx);
            }
        }
        __syncthreads();
        // ---- phase 2: h = tanh(T @ C^T + d) ----
        {
            const int n0 = w * 16;
            const bf16_t* brow = Cb + (n0 + rr) * RANK + kb;
            v16bf b0 = load_frag(brow, 0);
            v16bf b1 = load_frag(brow, 32);
#pragma unroll
            for (int mi = 0; mi < 4; ++mi) {
                const int m0 = mi * 16;
                const bf16_t* arow = T + (m0 + rr) * RANK + kb;
                v8f acc = {};
                acc = wmma_bf16(load_frag(arow, 0),  b0, acc);
                acc = wmma_bf16(load_frag(arow, 32), b1, acc);
                const int ncol = n0 + rr, mbase = m0 + mh;
#pragma unroll
                for (int r = 0; r < 8; ++r) {
                    int m = mbase + r;
                    float v = tanhf(acc[r] + dl[ncol]);
                    h[m * HIDDEN + ncol] = v;
                    Hb[((size_t)s * BATCH + m) * HIDDEN + ncol] = (bf16_t)v;
                }
            }
        }
        __syncthreads();
    }
    for (int i = tid; i < BATCH * HIDDEN; i += 1024) hout[i] = h[i];
}

// ---------------------------------------------------------------------------
// Kernel 5: decoder GEMM  logits[b][s][v] = Hb[s*64+b] . Wb[v] + b_dec[v]
//   M=16384 N=10000 K=512. Block = 128 threads (4 waves), 64x64 tile,
//   wave = 32x32 (2x2 WMMA tiles). Operands read straight from L2-resident
//   bf16 (both are per-lane-contiguous -> global_load_b128).
// ---------------------------------------------------------------------------
__global__ __launch_bounds__(128)
void k_decoder(const bf16_t* __restrict__ Hb, const bf16_t* __restrict__ Wb,
               const float* __restrict__ bdec, float* __restrict__ out) {
    const int w    = threadIdx.x >> 5;
    const int lane = threadIdx.x & 31;
    const int rr   = lane & 15;
    const int kb   = ((lane >> 4) & 1) * 8;
    const int m0   = blockIdx.y * 64 + (w >> 1) * 32;
    const int n0   = blockIdx.x * 64 + (w & 1) * 32;

    const bf16_t* arow0 = Hb + (size_t)(m0 + rr) * HIDDEN + kb;
    const bf16_t* arow1 = arow0 + (size_t)16 * HIDDEN;
    const int nrow0 = n0 + rr, nrow1 = nrow0 + 16;
    const bf16_t* brow0 = Wb + (size_t)nrow0 * HIDDEN + kb;
    const bf16_t* brow1 = Wb + (size_t)nrow1 * HIDDEN + kb;
    const bool bv0 = nrow0 < VOCAB;   // wave-uniform: VOCAB % 16 == 0
    const bool bv1 = nrow1 < VOCAB;

    v8f acc00 = {}, acc01 = {}, acc10 = {}, acc11 = {};
#pragma unroll 4
    for (int kt = 0; kt < HIDDEN / 32; ++kt) {
        const int k0 = kt * 32;
        __builtin_prefetch(arow0 + k0 + 64, 0, 3);
        __builtin_prefetch(brow0 + k0 + 64, 0, 3);
        v16bf a0 = load_frag(arow0, k0);
        v16bf a1 = load_frag(arow1, k0);
        v16bf b0 = zero_frag(), b1 = zero_frag();
        if (bv0) b0 = load_frag(brow0, k0);
        if (bv1) b1 = load_frag(brow1, k0);
        acc00 = wmma_bf16(a0, b0, acc00);
        acc01 = wmma_bf16(a0, b1, acc01);
        acc10 = wmma_bf16(a1, b0, acc10);
        acc11 = wmma_bf16(a1, b1, acc11);
    }

    const int mh = ((lane >> 4) & 1) ? 8 : 0;
#pragma unroll
    for (int j = 0; j < 2; ++j) {
        const int ncol = n0 + j * 16 + rr;
        if (ncol >= VOCAB) continue;
        const float bias = bdec[ncol];
#pragma unroll
        for (int i = 0; i < 2; ++i) {
            const v8f& acc = (i == 0) ? (j == 0 ? acc00 : acc01)
                                      : (j == 0 ? acc10 : acc11);
            const int mbase = m0 + i * 16 + mh;
#pragma unroll
            for (int r = 0; r < 8; ++r) {
                const int m = mbase + r;          // m = s*64 + b
                const int s = m >> 6, b = m & 63;
                out[(size_t)b * (SEQ * (size_t)VOCAB) + (size_t)s * VOCAB + ncol] =
                    acc[r] + bias;
            }
        }
    }
}

// ---------------------------------------------------------------------------
extern "C" void kernel_launch(void* const* d_in, const int* in_sizes, int n_in,
                              void* d_out, int out_size, void* d_ws, size_t ws_size,
                              hipStream_t stream) {
    (void)in_sizes; (void)n_in; (void)out_size; (void)ws_size;
    const int*   inp   = (const int*)  d_in[0];   // [BATCH][SEQ]
    const float* emb   = (const float*)d_in[1];   // [VOCAB][INPUT]
    const float* A     = (const float*)d_in[2];   // [HIDDEN][RANK]
    const float* B     = (const float*)d_in[3];   // [INPUT][RANK]
    const float* C     = (const float*)d_in[4];   // [HIDDEN][RANK]
    const float* dv    = (const float*)d_in[5];   // [HIDDEN]
    const float* W_dec = (const float*)d_in[6];   // [VOCAB][HIDDEN]
    const float* b_dec = (const float*)d_in[7];   // [VOCAB]
    float* out = (float*)d_out;                   // logits [B][S][V] ++ h_t [B][H]

    char* ws = (char*)d_ws;
    bf16_t* embb = (bf16_t*)(ws);                            // 5,120,000 B
    bf16_t* Wb   = (bf16_t*)(ws + 5120000);                  // 10,240,000 B
    bf16_t* Bt   = (bf16_t*)(ws + 15360000);                 // 32,768 B
    float*  Bx   = (float*) (ws + 15392768);                 // 4,194,304 B
    bf16_t* Hb   = (bf16_t*)(ws + 19587072);                 // 16,777,216 B

    k_f32_to_bf16<<<2048, 256, 0, stream>>>(emb,   embb, (long)VOCAB * INPUT);
    k_f32_to_bf16<<<2048, 256, 0, stream>>>(W_dec, Wb,   (long)VOCAB * HIDDEN);
    k_make_Bt<<<(RANK * INPUT + 255) / 256, 256, 0, stream>>>(B, Bt);

    k_bx<<<(SEQ * BATCH) / 64, 128, 0, stream>>>(inp, embb, Bt, Bx);

    const size_t lds_bytes = 131072 + 65536 + 65536 + 8192 + 2048; // 272,384
    float* hout = out + (size_t)BATCH * SEQ * VOCAB;
    k_recur<<<1, 1024, lds_bytes, stream>>>(A, C, dv, Bx, Hb, hout);

    dim3 dgrid((VOCAB + 63) / 64, (SEQ * BATCH) / 64);
    k_decoder<<<dgrid, 128, 0, stream>>>(Hb, Wb, b_dec, out);
}
